// Attn_23175643529485
// MI455X (gfx1250) — compile-verified, matrix operand order checked by reading, not run on hardware
//
#include <hip/hip_runtime.h>

// ---- problem constants (fixed by the reference harness) ----
#define BATCH   2
#define QLEN    2048
#define KLEN    2048
#define NHEADS  32
#define NKVH    8
#define HDIM    128
#define GROUP   (NHEADS / NKVH)   // 4
#define QPW     32                // queries per wave (2 x 16-query groups)
#define QPB     256               // queries per block (8 waves)

typedef __attribute__((ext_vector_type(16))) __bf16 v16bf;
typedef __attribute__((ext_vector_type(8)))  __bf16 v8bf;
typedef __attribute__((ext_vector_type(8)))  float  v8f;
typedef __attribute__((ext_vector_type(4)))  float  v4f;

// scale(1/sqrt(128)) folded with log2(e): softmax runs in exp2 domain
#define SCALE_LOG2E 0.1275476323540085f

// Cooperative flash attention, wave32, gfx1250 WMMA. Final configuration.
//  - Workgroup = 8 waves, 256 queries of one (batch, head); 512 blocks.
//  - Per 32-key step: block stages K (row-major bf16) and Vᵀ (dim-major bf16)
//    into LDS once; waves consume fragments via contiguous ds_load_b128.
//  - Each wave owns TWO 16-query groups so every K/Vᵀ fragment feeds 2 WMMAs
//    (halves LDS bytes per matrix op — the binding roofline at this tiling).
//  - ~300 VGPRs/lane, register-resident accumulators: uses CDNA5's 1024-deep
//    VGPR file via MSB addressing; s_set_vgpr_msb SALU prefixes are near-free
//    on the scalar pipe, whereas capping at 256 regs provably spilled (~37
//    slots -> scratch traffic every key-step). No spills in this config.
//  - Sᵀ = K·Qᵀ: per-query softmax reduce = 8 in-lane values + 1 shfl_xor(16),
//    and Pᵀ's Sᵀ C-layout IS the B-fragment layout of Oᵀ = Vᵀ·Pᵀ (no shuffles).
//  - Exactly one diagonal (masked) step per wave; mask-free fast path else.
//  - Zero-register pipelining: each thread global_prefetch's one 64B line of
//    next-step K and V right after the barrier; WMMAs hide the cache fill.
__global__ __launch_bounds__(256, 2)
void fa_fwd_gqa_causal(const float* __restrict__ q,
                       const float* __restrict__ k,
                       const float* __restrict__ v,
                       float* __restrict__ out)
{
    __shared__ __align__(16) __bf16 lk[32 * HDIM];   // K tile  [key][dim]  8 KB
    __shared__ __align__(16) __bf16 lv[HDIM * 32];   // Vᵀ tile [dim][key]  8 KB

    const int tid  = threadIdx.x;
    const int lane = tid & 31;
    const int wave = tid >> 5;

    const int qblks = QLEN / QPB;                  // 8 query-blocks per head
    const int bi    = blockIdx.x;
    const int b     = bi / (NHEADS * qblks);
    const int rem   = bi % (NHEADS * qblks);
    const int h     = rem / qblks;
    const int qblk  = rem % qblks;
    const int bq0   = qblk * QPB;
    const int q0    = bq0 + wave * QPW;            // wave's first query (32-aligned)
    const int kvh   = h / GROUP;

    const int half = lane >> 4;
    const int qcol = lane & 15;
    const int kb   = half * 8;

    // staging roles (also used for next-tile line prefetch)
    const int k_key  = tid >> 3;                   // one key, 16-dim segment
    const int k_dseg = (tid & 7) * 16;
    const int v_dim  = tid & 127;                  // one dim, 16 keys
    const int v_kg   = (tid >> 7) * 16;

    const size_t kstride = (size_t)NKVH * HDIM;
    const float* kbase = k + ((size_t)b * KLEN * NKVH + kvh) * HDIM;
    const float* vbase = v + ((size_t)b * KLEN * NKVH + kvh) * HDIM;

    // ---- Qᵀ B-fragments for both 16-query groups ----
    v16bf qb[2][4];
#pragma unroll
    for (int g = 0; g < 2; ++g) {
        const float* qrow = q + (((size_t)b * QLEN + q0 + g * 16 + qcol) * NHEADS + h) * HDIM;
#pragma unroll
        for (int c = 0; c < 4; ++c) {
            v4f f0 = *(const v4f*)(qrow + c * 32 + kb);
            v4f f1 = *(const v4f*)(qrow + c * 32 + kb + 4);
            v4f f2 = *(const v4f*)(qrow + c * 32 + 16 + kb);
            v4f f3 = *(const v4f*)(qrow + c * 32 + 16 + kb + 4);
#pragma unroll
            for (int e = 0; e < 4; ++e) {
                qb[g][c][e]      = (__bf16)f0[e];
                qb[g][c][4 + e]  = (__bf16)f1[e];
                qb[g][c][8 + e]  = (__bf16)f2[e];
                qb[g][c][12 + e] = (__bf16)f3[e];
            }
        }
    }

    // ---- softmax state per query group (exp2 domain) ----
    float m[2]  = { -__builtin_inff(), -__builtin_inff() };
    float ls[2] = { 0.0f, 0.0f };
    v8f o[2][8];
#pragma unroll
    for (int g = 0; g < 2; ++g)
#pragma unroll
        for (int dc = 0; dc < 8; ++dc) o[g][dc] = (v8f){};

    const int nsteps = bq0 / 32 + QPB / 32;        // keys 0 .. bq0+255

    for (int step = 0; step < nsteps; ++step) {
        const int j0 = step * 32;

        // ---- stage this tile: global f32 -> bf16 LDS ----
        {
            const float* kr = kbase + (size_t)(j0 + k_key) * kstride + k_dseg;
            v4f a0 = *(const v4f*)(kr + 0);
            v4f a1 = *(const v4f*)(kr + 4);
            v4f a2 = *(const v4f*)(kr + 8);
            v4f a3 = *(const v4f*)(kr + 12);
            v8bf r0, r1;
#pragma unroll
            for (int e = 0; e < 4; ++e) {
                r0[e]     = (__bf16)a0[e];
                r0[4 + e] = (__bf16)a1[e];
                r1[e]     = (__bf16)a2[e];
                r1[4 + e] = (__bf16)a3[e];
            }
            *(v8bf*)&lk[k_key * HDIM + k_dseg]     = r0;
            *(v8bf*)&lk[k_key * HDIM + k_dseg + 8] = r1;

            const float* vr = vbase + (size_t)(j0 + v_kg) * kstride + v_dim;
            v8bf t0, t1;
#pragma unroll
            for (int e = 0; e < 8; ++e) {
                t0[e] = (__bf16)vr[(size_t)e * kstride];
                t1[e] = (__bf16)vr[(size_t)(8 + e) * kstride];
            }
            *(v8bf*)&lv[v_dim * 32 + v_kg]     = t0;
            *(v8bf*)&lv[v_dim * 32 + v_kg + 8] = t1;
        }
        __syncthreads();

        // ---- zero-register pipeline: prefetch next tile's cachelines ----
        // K and V tiles each consist of 256 distinct 64B lines; the 256 threads
        // cover them 1:1 using the K staging role. WMMAs below hide the fill.
        {
            const int pj0 = (step + 1 < nsteps) ? (j0 + 32) : j0;
            const float* pk = kbase + (size_t)(pj0 + k_key) * kstride + k_dseg;
            const float* pv = vbase + (size_t)(pj0 + k_key) * kstride + k_dseg;
            __builtin_prefetch(pk, 0, 1);
            __builtin_prefetch(pv, 0, 1);
        }

        // ---- per-wave compute (skip once fully above the diagonal) ----
        if (j0 < q0 + QPW) {
            // Sᵀ = K · Qᵀ : two 16-key subtiles x two query groups
            v8f st[2][2];
#pragma unroll
            for (int s = 0; s < 2; ++s) {
                st[0][s] = (v8f){};
                st[1][s] = (v8f){};
                const __bf16* krow = &lk[(s * 16 + qcol) * HDIM];
#pragma unroll
                for (int c = 0; c < 4; ++c) {
                    v8bf lo = *(const v8bf*)(krow + c * 32 + kb);
                    v8bf hi = *(const v8bf*)(krow + c * 32 + 16 + kb);
                    v16bf ka = __builtin_shufflevector(lo, hi,
                        0, 1, 2, 3, 4, 5, 6, 7, 8, 9, 10, 11, 12, 13, 14, 15);
                    st[0][s] = __builtin_amdgcn_wmma_f32_16x16x32_bf16(
                                   false, ka, false, qb[0][c], (short)0, st[0][s], false, false);
                    st[1][s] = __builtin_amdgcn_wmma_f32_16x16x32_bf16(
                                   false, ka, false, qb[1][c], (short)0, st[1][s], false, false);
                }
            }

            // scale (+ mask only on the single diagonal step), per-group max
            float mloc[2];
            if (j0 + 32 > q0) {                    // diagonal step (exactly one per wave)
#pragma unroll
                for (int g = 0; g < 2; ++g) {
                    const int i_q = q0 + g * 16 + qcol;
                    mloc[g] = -__builtin_inff();
#pragma unroll
                    for (int s = 0; s < 2; ++s)
#pragma unroll
                        for (int r = 0; r < 8; ++r) {
                            const int j = j0 + s * 16 + r + kb;
                            float val = st[g][s][r] * SCALE_LOG2E;
                            val = (j > i_q) ? -__builtin_inff() : val;
                            st[g][s][r] = val;
                            mloc[g] = fmaxf(mloc[g], val);
                        }
                }
            } else {                               // mask-free fast path
#pragma unroll
                for (int g = 0; g < 2; ++g) {
                    mloc[g] = -__builtin_inff();
#pragma unroll
                    for (int s = 0; s < 2; ++s)
#pragma unroll
                        for (int r = 0; r < 8; ++r) {
                            const float val = st[g][s][r] * SCALE_LOG2E;
                            st[g][s][r] = val;
                            mloc[g] = fmaxf(mloc[g], val);
                        }
                }
            }

            // online softmax update (exp2 domain) + Pᵀ B-fragments
            v16bf pb[2];
            float alpha[2];
#pragma unroll
            for (int g = 0; g < 2; ++g) {
                mloc[g] = fmaxf(mloc[g], __shfl_xor(mloc[g], 16, 32));
                const float mnew = fmaxf(m[g], mloc[g]);
                alpha[g] = __builtin_amdgcn_exp2f(m[g] - mnew);
                float psum = 0.0f;
#pragma unroll
                for (int s = 0; s < 2; ++s)
#pragma unroll
                    for (int r = 0; r < 8; ++r) {
                        const float p = __builtin_amdgcn_exp2f(st[g][s][r] - mnew);
                        psum += p;
                        pb[g][s * 8 + r] = (__bf16)p;
                    }
                psum += __shfl_xor(psum, 16, 32);
                ls[g] = ls[g] * alpha[g] + psum;
                m[g]  = mnew;
            }

            // Oᵀ += Vᵀ·Pᵀ : each Vᵀ fragment feeds both query groups
#pragma unroll
            for (int dc = 0; dc < 8; ++dc) {
                const __bf16* vrow = &lv[(dc * 16 + qcol) * 32];
                v8bf lo = *(const v8bf*)(vrow + kb);
                v8bf hi = *(const v8bf*)(vrow + 16 + kb);
                v16bf va = __builtin_shufflevector(lo, hi,
                    0, 1, 2, 3, 4, 5, 6, 7, 8, 9, 10, 11, 12, 13, 14, 15);
#pragma unroll
                for (int g = 0; g < 2; ++g) {
                    v8f oo;
#pragma unroll
                    for (int r = 0; r < 8; ++r) oo[r] = o[g][dc][r] * alpha[g];
                    o[g][dc] = __builtin_amdgcn_wmma_f32_16x16x32_bf16(
                                   false, va, false, pb[g], (short)0, oo, false, false);
                }
            }
        }
        __syncthreads();                           // LDS safe to overwrite
    }

    // ---- epilogue: O[q][d] = Oᵀ[d][q] / l ----
#pragma unroll
    for (int g = 0; g < 2; ++g) {
        const float inv = 1.0f / ls[g];
        float* orow = out + (((size_t)b * QLEN + q0 + g * 16 + qcol) * NHEADS + h) * HDIM;
#pragma unroll
        for (int dc = 0; dc < 8; ++dc)
#pragma unroll
            for (int r = 0; r < 8; ++r) {
                const int d = dc * 16 + r + kb;
                orow[d] = o[g][dc][r] * inv;
            }
    }
}

extern "C" void kernel_launch(void* const* d_in, const int* in_sizes, int n_in,
                              void* d_out, int out_size, void* d_ws, size_t ws_size,
                              hipStream_t stream)
{
    (void)in_sizes; (void)n_in; (void)out_size; (void)d_ws; (void)ws_size;
    const float* q = (const float*)d_in[0];
    const float* k = (const float*)d_in[1];
    const float* v = (const float*)d_in[2];
    float* out = (float*)d_out;

    const int blocks = BATCH * NHEADS * (QLEN / QPB);  // 512 workgroups
    fa_fwd_gqa_causal<<<blocks, 256, 0, stream>>>(q, k, v, out);
}
